// MLP_35708358099485
// MI455X (gfx1250) — matrix operand order repaired and checked by reading, not compiled, not run on hardware
//
#include <hip/hip_runtime.h>
#include <stdint.h>

// ---------------------------------------------------------------------------
// Types for CDNA5 WMMA (wave32): bf16 operands, f32 accumulator.
// ---------------------------------------------------------------------------
typedef __bf16 bf16_t;
typedef bf16_t v16bf __attribute__((ext_vector_type(16)));
typedef bf16_t v8bf  __attribute__((ext_vector_type(8)));
typedef float  v8f   __attribute__((ext_vector_type(8)));
typedef int    v4i   __attribute__((ext_vector_type(4)));

__device__ __forceinline__ unsigned short f32_to_bf16(float f) {
    unsigned u = __float_as_uint(f);
    u += 0x7fffu + ((u >> 16) & 1u);   // round-to-nearest-even
    return (unsigned short)(u >> 16);
}

// ---------------------------------------------------------------------------
// Async global->LDS B128 copy (ASYNCcnt path, CDNA5 ISA).
// Builtin signature (from probe diagnostics): (v4i AS1*, v4i AS3*, Imm, Imm).
// Generic LDS pointers carry the LDS byte offset in their low 32 bits
// (LDS aperture rule); integer casts avoid pointer
// address-space-cast legality questions.
// ---------------------------------------------------------------------------
__device__ __forceinline__ void async_copy_b128(const void* g, void* l) {
#if __has_builtin(__builtin_amdgcn_global_load_async_to_lds_b128)
    __builtin_amdgcn_global_load_async_to_lds_b128(
        (__attribute__((address_space(1))) v4i*)(uintptr_t)g,
        (__attribute__((address_space(3))) v4i*)(uint32_t)(uintptr_t)l,
        0, 0);
#else
    uint32_t lds_off = (uint32_t)(uintptr_t)l;
    asm volatile("global_load_async_to_lds_b128 %0, %1, off"
                 :: "v"(lds_off), "v"((uint64_t)(uintptr_t)g)
                 : "memory");
#endif
}

__device__ __forceinline__ void wait_async_le8() {
    asm volatile("s_wait_asynccnt 0x8" ::: "memory");
}
__device__ __forceinline__ void wait_async_0() {
    asm volatile("s_wait_asynccnt 0x0" ::: "memory");
}

// ---------------------------------------------------------------------------
// Global top-k threshold selection (exact, two-level radix on |s| bits).
// ---------------------------------------------------------------------------
__global__ void zero_u32_kernel(unsigned* __restrict__ p, int n) {
    int i = blockIdx.x * blockDim.x + threadIdx.x;
    int st = gridDim.x * blockDim.x;
    for (; i < n; i += st) p[i] = 0u;
}

__global__ void hist_hi_kernel(const float* __restrict__ s, int n,
                               unsigned* __restrict__ hist) {
    int i = blockIdx.x * blockDim.x + threadIdx.x;
    int st = gridDim.x * blockDim.x;
    for (; i < n; i += st) {
        unsigned u = __float_as_uint(s[i]) & 0x7fffffffu;
        atomicAdd(&hist[u >> 16], 1u);
    }
}

__global__ void hist_lo_kernel(const float* __restrict__ s, int n,
                               const unsigned* __restrict__ resHi,
                               unsigned* __restrict__ hist) {
    const unsigned H = resHi[0];
    int i = blockIdx.x * blockDim.x + threadIdx.x;
    int st = gridDim.x * blockDim.x;
    for (; i < n; i += st) {
        unsigned u = __float_as_uint(s[i]) & 0x7fffffffu;
        if ((u >> 16) == H) atomicAdd(&hist[u & 0xffffu], 1u);
    }
}

// Finds bin b = min{b : cum(b) > target} over a 65536-bin histogram.
// res[0] = bin, res[1] = target - (count strictly below bin).
__global__ void scan_hist_kernel(const unsigned* __restrict__ hist,
                                 unsigned target_imm,
                                 const unsigned* __restrict__ target_ptr,
                                 unsigned* __restrict__ res) {
    __shared__ unsigned part[256];
    const unsigned t = threadIdx.x;
    unsigned sum = 0;
    for (int i = 0; i < 256; ++i) sum += hist[t * 256 + i];
    part[t] = sum;
    __syncthreads();
    if (t == 0) {
        unsigned target = target_ptr ? target_ptr[1] : target_imm;
        unsigned cum = 0;
        int seg = 0;
        for (; seg < 256; ++seg) {
            if (cum + part[seg] > target) break;
            cum += part[seg];
        }
        if (seg == 256) seg = 255;
        unsigned bin = (unsigned)seg * 256u;
        for (int i = 0; i < 256; ++i) {
            unsigned c = hist[seg * 256 + i];
            if (cum + c > target) { bin = (unsigned)seg * 256u + i; break; }
            cum += c;
        }
        res[0] = bin;
        res[1] = target - cum;
    }
}

// ---------------------------------------------------------------------------
// Pack kernels: apply mask (|s| bits >= threshold) and convert to bf16.
// ---------------------------------------------------------------------------
__global__ void pack_masked_kernel(const float* __restrict__ w,
                                   const float* __restrict__ s,
                                   const unsigned* __restrict__ resHi,
                                   const unsigned* __restrict__ resLo,
                                   unsigned short* __restrict__ out, int n) {
    const unsigned T = (resHi[0] << 16) | resLo[0];
    int i = blockIdx.x * blockDim.x + threadIdx.x;
    int st = gridDim.x * blockDim.x;
    for (; i < n; i += st) {
        unsigned u = __float_as_uint(s[i]) & 0x7fffffffu;
        float v = (u >= T) ? w[i] : 0.0f;
        out[i] = f32_to_bf16(v);
    }
}

__global__ void pack_kernel(const float* __restrict__ x,
                            unsigned short* __restrict__ out, int n) {
    int i = blockIdx.x * blockDim.x + threadIdx.x;
    int st = gridDim.x * blockDim.x;
    for (; i < n; i += st) out[i] = f32_to_bf16(x[i]);
}

// ---------------------------------------------------------------------------
// WMMA GEMM with async-to-LDS double buffering.
//   C[m,o] = sum_k A[m,k]*B[o,k]  (+ bias[o], optional ReLU)
//   A: M x K bf16 row-major (activations), B: N x K bf16 row-major (weights).
// Block tile 128x128, Ktile 64, 8 wave32s, each wave 64x32 = 4x2 WMMAs.
// LDS stage: A 128x64 + B 128x64 bf16, rows padded to 144 B (36-dword
// stride -> the 16 rows touched by a half-wave fragment read map to 16
// distinct banks). 2 stages = 73728 B dynamic LDS (4 WGs / 320KB WGP).
// Each wave issues 8 async B128 copies per stage; with the next stage in
// flight, ASYNCcnt<=8 (in-order completion) proves the current stage landed.
// ---------------------------------------------------------------------------
#define TILE_K      64
#define LDS_ROW_B   144                        // 64*2 + 16 pad
#define LDS_A_BYTES (128 * LDS_ROW_B)          // 18432
#define LDS_STAGE   (2 * LDS_A_BYTES)          // 36864 (A then B)

template <bool RELU, bool BF16OUT>
__global__ __launch_bounds__(256) void gemm_wmma_kernel(
    const unsigned short* __restrict__ A,
    const unsigned short* __restrict__ B,
    const float* __restrict__ bias,
    void* __restrict__ C,
    int Kdim, int ldc, int Nvalid) {
    extern __shared__ char smem[];

    const int tid      = threadIdx.x;
    const int lane     = tid & 31;
    const int wave     = tid >> 5;
    const int wm       = wave >> 2;          // 0..1  (wave row)
    const int wn       = wave & 3;           // 0..3  (wave col)
    const int blockM   = blockIdx.y * 128;
    const int blockN   = blockIdx.x * 128;
    const int laneHalf = lane >> 4;          // 0 or 1
    const int laneMod  = lane & 15;

    v8f acc[4][2];
    const v8f vzero = {0.f, 0.f, 0.f, 0.f, 0.f, 0.f, 0.f, 0.f};
#pragma unroll
    for (int i = 0; i < 4; ++i)
#pragma unroll
        for (int j = 0; j < 2; ++j) acc[i][j] = vzero;

    // ---- async stage issue: 1024 16B chunks per matrix, 4 per thread ----
    auto issue_stage = [&](int kt, int s) {
        char* As = smem + s * LDS_STAGE;
        char* Bs = As + LDS_A_BYTES;
        const char* Ag = (const char*)A + (size_t)kt * TILE_K * 2;
        const char* Bg = (const char*)B + (size_t)kt * TILE_K * 2;
#pragma unroll
        for (int i = 0; i < 4; ++i) {
            int c   = tid + i * 256;         // 0..1023
            int row = c >> 3;                // 0..127
            int cb  = (c & 7) * 16;          // byte col within 128B row
            async_copy_b128(Ag + (size_t)(blockM + row) * Kdim * 2 + cb,
                            As + row * LDS_ROW_B + cb);
        }
#pragma unroll
        for (int i = 0; i < 4; ++i) {
            int c   = tid + i * 256;
            int row = c >> 3;
            int rg  = min(blockN + row, Nvalid - 1);   // clamp ragged N
            int cb  = (c & 7) * 16;
            async_copy_b128(Bg + (size_t)rg * Kdim * 2 + cb,
                            Bs + row * LDS_ROW_B + cb);
        }
    };

    const int nkt = Kdim / TILE_K;
    issue_stage(0, 0);

    for (int kt = 0; kt < nkt; ++kt) {
        const int cur = kt & 1;
        if (kt + 1 < nkt) {
            issue_stage(kt + 1, cur ^ 1);    // prev reads of that buffer were
            wait_async_le8();                // fenced by last iter's barrier
        } else {
            wait_async_0();
        }
        __syncthreads();                     // stage `cur` fully in LDS

        const char* As = smem + cur * LDS_STAGE;
        const char* Bs = As + LDS_A_BYTES;
#pragma unroll
        for (int k0 = 0; k0 < TILE_K; k0 += 32) {
            v16bf bfrag[2];
#pragma unroll
            for (int j = 0; j < 2; ++j) {
                const char* p = Bs + (wn * 32 + j * 16 + laneMod) * LDS_ROW_B
                              + (k0 + laneHalf * 16) * 2;
                union { v16bf v; v8bf h[2]; } u;
                u.h[0] = *reinterpret_cast<const v8bf*>(p);        // K[+0..7]
                u.h[1] = *reinterpret_cast<const v8bf*>(p + 16);   // K[+8..15]
                bfrag[j] = u.v;
            }
            v16bf afrag[4];
#pragma unroll
            for (int i = 0; i < 4; ++i) {
                const char* p = As + (wm * 64 + i * 16 + laneMod) * LDS_ROW_B
                              + (k0 + laneHalf * 16) * 2;
                union { v16bf v; v8bf h[2]; } u;
                u.h[0] = *reinterpret_cast<const v8bf*>(p);        // K[+0..7]
                u.h[1] = *reinterpret_cast<const v8bf*>(p + 16);   // K[+8..15]
                afrag[i] = u.v;
            }
#pragma unroll
            for (int i = 0; i < 4; ++i)
#pragma unroll
                for (int j = 0; j < 2; ++j)
                    acc[i][j] = __builtin_amdgcn_wmma_f32_16x16x32_bf16(
                        false, afrag[i], false, bfrag[j],
                        (short)0, acc[i][j], false, false);
        }
        __syncthreads();                     // done reading `cur` before reuse
    }

    // Epilogue: bias (+ReLU), store per the 16x16 f32 C/D layout.
#pragma unroll
    for (int i = 0; i < 4; ++i) {
#pragma unroll
        for (int j = 0; j < 2; ++j) {
            const int tm   = blockM + wm * 64 + i * 16;
            const int tn   = blockN + wn * 32 + j * 16;
            const int col  = tn + laneMod;
            const int rb   = tm + (laneHalf << 3);
            const bool ok  = (col < Nvalid);
            const float bv = bias[min(col, Nvalid - 1)];
#pragma unroll
            for (int r = 0; r < 8; ++r) {
                float v = acc[i][j][r] + bv;
                if (RELU) v = fmaxf(v, 0.0f);
                if (ok) {
                    if (BF16OUT)
                        ((unsigned short*)C)[(size_t)(rb + r) * ldc + col] = f32_to_bf16(v);
                    else
                        ((float*)C)[(size_t)(rb + r) * ldc + col] = v;
                }
            }
        }
    }
}

// ---------------------------------------------------------------------------
// Launch
// ---------------------------------------------------------------------------
extern "C" void kernel_launch(void* const* d_in, const int* in_sizes, int n_in,
                              void* d_out, int out_size, void* d_ws, size_t ws_size,
                              hipStream_t stream) {
    (void)in_sizes; (void)n_in; (void)out_size; (void)ws_size;
    const float* x  = (const float*)d_in[0];
    const float* w1 = (const float*)d_in[1];
    const float* s1 = (const float*)d_in[2];
    const float* b1 = (const float*)d_in[3];
    const float* w2 = (const float*)d_in[4];
    const float* s2 = (const float*)d_in[5];
    const float* b2 = (const float*)d_in[6];
    const float* w3 = (const float*)d_in[7];
    const float* s3 = (const float*)d_in[8];
    const float* b3 = (const float*)d_in[9];
    float* out = (float*)d_out;

    const int Bsz = 4096, I0 = 2048, I1 = 4096, I2 = 4096, O = 1000;

    // Workspace carve-out (256B aligned): ~143 MB total.
    char* ws = (char*)d_ws;
    size_t off = 0;
    auto carve = [&](size_t bytes) -> char* {
        char* p = ws + off;
        off = (off + bytes + 255) & ~(size_t)255;
        return p;
    };
    unsigned short* Xb  = (unsigned short*)carve((size_t)Bsz * I0 * 2);
    unsigned short* W1b = (unsigned short*)carve((size_t)I1 * I0 * 2);
    unsigned short* W2b = (unsigned short*)carve((size_t)I2 * I1 * 2);
    unsigned short* W3b = (unsigned short*)carve((size_t)O * I2 * 2);
    unsigned short* H1  = (unsigned short*)carve((size_t)Bsz * I1 * 2);
    unsigned short* H2  = (unsigned short*)carve((size_t)Bsz * I2 * 2);
    unsigned* histA = (unsigned*)carve(65536 * 4);
    unsigned* histB = (unsigned*)carve(65536 * 4);
    unsigned* resA  = (unsigned*)carve(64);
    unsigned* resB  = (unsigned*)carve(64);

    auto select_and_pack = [&](const float* w, const float* s, int n,
                               unsigned short* outw) {
        zero_u32_kernel<<<256, 256, 0, stream>>>(histA, 65536);
        zero_u32_kernel<<<256, 256, 0, stream>>>(histB, 65536);
        hist_hi_kernel<<<2048, 256, 0, stream>>>(s, n, histA);
        // rank j = (1-K)*n = n/2 smallest get zeroed; threshold = value at rank j
        scan_hist_kernel<<<1, 256, 0, stream>>>(histA, (unsigned)(n / 2), nullptr, resA);
        hist_lo_kernel<<<2048, 256, 0, stream>>>(s, n, resA, histB);
        scan_hist_kernel<<<1, 256, 0, stream>>>(histB, 0u, resA, resB);
        pack_masked_kernel<<<2048, 256, 0, stream>>>(w, s, resA, resB, outw, n);
    };

    select_and_pack(w1, s1, I1 * I0, W1b);
    select_and_pack(w2, s2, I2 * I1, W2b);
    select_and_pack(w3, s3, O * I2, W3b);
    pack_kernel<<<2048, 256, 0, stream>>>(x, Xb, Bsz * I0);

    const size_t lds_bytes = 2 * LDS_STAGE;  // 73728
    dim3 blk(256);
    gemm_wmma_kernel<true, true><<<dim3(I1 / 128, Bsz / 128), blk, lds_bytes, stream>>>(
        Xb, W1b, b1, H1, I0, I1, I1);
    gemm_wmma_kernel<true, true><<<dim3(I2 / 128, Bsz / 128), blk, lds_bytes, stream>>>(
        H1, W2b, b2, H2, I1, I2, I2);
    gemm_wmma_kernel<false, false><<<dim3((O + 127) / 128, Bsz / 128), blk, lds_bytes, stream>>>(
        H2, W3b, b3, out, I2, O, O);
}
